// GraphConv_61675730370843
// MI455X (gfx1250) — compile-verified, matrix operand order checked by reading, not run on hardware
//
#include <hip/hip_runtime.h>
#include <hip/hip_bf16.h>
#include <stdint.h>

// Shapes from the reference
#define B_  8
#define N_  8192
#define C_  32
#define S_  2
#define KT_ 256   // N_/32 k-steps of the bf16 WMMA

typedef __attribute__((ext_vector_type(16))) __bf16 v16bf;
typedef __attribute__((ext_vector_type(2)))  __bf16 v2bf;
typedef __attribute__((ext_vector_type(8)))  float  v8f;
typedef __attribute__((ext_vector_type(4)))  float  v4f;   // clang vector: OK for nontemporal builtins
typedef __attribute__((ext_vector_type(2)))  float  v2f;

union Frag16 { uint4 u4[2]; uint32_t u[8]; v16bf v; };

// pack two fp32 into one dword of two bf16
__device__ __forceinline__ uint32_t pkbf(float a, float b) {
#if __has_builtin(__builtin_amdgcn_cvt_pk_bf16_f32)
  v2bf p = __builtin_amdgcn_cvt_pk_bf16_f32(a, b);   // 1 VALU op, RNE
  return __builtin_bit_cast(uint32_t, p);
#else
  return (__float_as_uint(a) >> 16) | (__float_as_uint(b) & 0xFFFF0000u);
#endif
}

// ---------------------------------------------------------------------------
// Pre-pass: convert x (B,N,C_IN) fp32 -> bf16 WMMA B-operand fragments.
// Fragment id = (b*KT_ + kt)*2 + half  (half: C_IN cols 0-15 / 16-31).
// Per fragment: 32 lanes x 8 dwords (v16bf per lane) => 2 x b128 per lane.
// 16-bit B 32x16 layout (mirror of the A layout):
//   lanes 0-15 : col = lane,    K = 0..7 (pairs in dw0..3) and 16..23 (dw4..7)
//   lanes16-31 : col = lane-16, K = 8..15            and 24..31
// ---------------------------------------------------------------------------
__global__ void pack_x_kernel(const float* __restrict__ x,
                              uint32_t* __restrict__ xp) {
  int tid  = blockIdx.x * blockDim.x + threadIdx.x;
  int lane = tid & 31;
  int frag = tid >> 5;
  if (frag >= B_ * KT_ * 2) return;
  int half = frag & 1;
  int kt   = (frag >> 1) & (KT_ - 1);
  int b    = frag >> 9;                       // frag / (KT_*2)
  int col  = (lane & 15) + half * 16;
  int k0   = kt * 32 + ((lane >> 4) << 3);    // +8 for upper half-wave
  uint32_t o[8];
#pragma unroll
  for (int ch = 0; ch < 2; ++ch) {            // K chunk +0 / +16
#pragma unroll
    for (int p = 0; p < 4; ++p) {
      int k = k0 + ch * 16 + p * 2;
      float a = x[((size_t)b * N_ + k)     * C_ + col];
      float c = x[((size_t)b * N_ + k + 1) * C_ + col];
      o[ch * 4 + p] = pkbf(a, c);
    }
  }
  uint32_t* dst = xp + (size_t)frag * 256 + lane * 8;
  *(uint4*)(dst)     = make_uint4(o[0], o[1], o[2], o[3]);
  *(uint4*)(dst + 4) = make_uint4(o[4], o[5], o[6], o[7]);
}

// ---------------------------------------------------------------------------
// Main kernel: one wave per 16-row tile of out. For each s: stream 16 rows of
// supports[s] exactly once (non-temporal b128 loads -> bf16 WMMA, K=8192),
// keep h accumulators for all 8 batches x 2 col-halves in VGPRs (128 VGPRs),
// then apply W[s] in full fp32 precision (v_wmma_f32_16x16x4_f32) via a
// per-wave LDS re-layout and accumulate into out.
// ---------------------------------------------------------------------------
__global__ void __launch_bounds__(256, 1)
graphconv_kernel(const float* __restrict__ sup, const uint32_t* __restrict__ xp,
                 const float* __restrict__ W,   const float* __restrict__ bias,
                 float* __restrict__ out) {
  __shared__ float lds_h[8 * 16 * 32];            // per-wave 16x32 fp32 h tile
  const int lane = threadIdx.x & 31;
  const int wave = threadIdx.x >> 5;
  const int tile = blockIdx.x * 8 + wave;         // 0..511
  const int row0 = tile * 16;
  const int mrow = lane & 15;
  const int hi   = lane >> 4;                     // upper half-wave flag
  float* hb = lds_h + wave * 512;

  const v8f vzero = {0.f, 0.f, 0.f, 0.f, 0.f, 0.f, 0.f, 0.f};

#pragma unroll 1
  for (int s = 0; s < S_; ++s) {
    const float* arow = sup + (size_t)s * N_ * N_ + (size_t)(row0 + mrow) * N_;

    v8f acc[B_][2];                               // h accumulators
#pragma unroll
    for (int b = 0; b < B_; ++b) { acc[b][0] = vzero; acc[b][1] = vzero; }

#pragma unroll 2
    for (int kt = 0; kt < KT_; ++kt) {
      const int k0 = kt * 32 + hi * 8;
      // pull the supports stream ahead into near cache (WGP-scope prefetch)
      __builtin_prefetch(arow + k0 + 4 * 32, 0, 3);

      // A fragment: 16x32 bf16, converted from fp32 in registers.
      // supports is read-once: non-temporal loads keep it from evicting the
      // hot packed-x fragments out of WGP$/L2.
      Frag16 A;
      {
        v4f f0 = __builtin_nontemporal_load((const v4f*)(arow + k0));
        v4f f1 = __builtin_nontemporal_load((const v4f*)(arow + k0 + 4));
        v4f f2 = __builtin_nontemporal_load((const v4f*)(arow + k0 + 16));
        v4f f3 = __builtin_nontemporal_load((const v4f*)(arow + k0 + 20));
        A.u[0] = pkbf(f0.x, f0.y); A.u[1] = pkbf(f0.z, f0.w);
        A.u[2] = pkbf(f1.x, f1.y); A.u[3] = pkbf(f1.z, f1.w);
        A.u[4] = pkbf(f2.x, f2.y); A.u[5] = pkbf(f2.z, f2.w);
        A.u[6] = pkbf(f3.x, f3.y); A.u[7] = pkbf(f3.z, f3.w);
      }

      const uint32_t* xb = xp + (size_t)kt * 2 * 256 + lane * 8;
#pragma unroll
      for (int b = 0; b < B_; ++b) {
        const uint32_t* p0 = xb + (size_t)b * (KT_ * 2) * 256;
        Frag16 B0, B1;
        B0.u4[0] = ((const uint4*)p0)[0];
        B0.u4[1] = ((const uint4*)p0)[1];
        B1.u4[0] = ((const uint4*)(p0 + 256))[0];
        B1.u4[1] = ((const uint4*)(p0 + 256))[1];
        acc[b][0] = __builtin_amdgcn_wmma_f32_16x16x32_bf16(
            false, A.v, false, B0.v, (short)0, acc[b][0], false, false);
        acc[b][1] = __builtin_amdgcn_wmma_f32_16x16x32_bf16(
            false, A.v, false, B1.v, (short)0, acc[b][1], false, false);
      }
    }

    // ---- epilogue: out[b, row0+m, o] (+)= sum_c h[m,c] * W[s,o,c] ----------
#pragma unroll 1
    for (int b = 0; b < B_; ++b) {
      // spill h tile (C-layout) to per-wave LDS in plain [m][c] addressing
#pragma unroll
      for (int hh = 0; hh < 2; ++hh)
#pragma unroll
        for (int j = 0; j < 8; ++j)
          hb[(j + 8 * hi) * 32 + mrow + 16 * hh] = acc[b][hh][j];

#pragma unroll
      for (int oh = 0; oh < 2; ++oh) {
        v8f d = vzero;                            // D(o,m) 16x16 accumulator
#pragma unroll
        for (int cb = 0; cb < 32; cb += 4) {      // fp32 WMMA, K=4 per step
          // A = W[s] chunk: (o = lane%16 within oh tile, k = cb + hi*2 + {0,1})
          const float* wp = W + ((size_t)s * 32 + oh * 16 + mrow) * 32 + cb + hi * 2;
          v2f a2; a2.x = wp[0]; a2.y = wp[1];
          // B = h^T chunk from LDS: (k = cb + hi*2 + {0,1}, n = m = lane%16)
          const float* hp = hb + mrow * 32 + cb + hi * 2;
          v2f b2; b2.x = hp[0]; b2.y = hp[1];
          d = __builtin_amdgcn_wmma_f32_16x16x4_f32(
              false, a2, false, b2, (short)0, d, false, false);
        }
#pragma unroll
        for (int j = 0; j < 8; ++j) {
          int o = oh * 16 + j + 8 * hi;
          size_t idx = ((size_t)b * N_ + row0 + mrow) * C_ + o;
          if (s == 0) out[idx] = d[j] + bias[o];  // first pass: write (+bias)
          else        out[idx] += d[j];           // second s: accumulate
        }
      }
    }
  }
}

// ---------------------------------------------------------------------------
extern "C" void kernel_launch(void* const* d_in, const int* in_sizes, int n_in,
                              void* d_out, int out_size, void* d_ws, size_t ws_size,
                              hipStream_t stream) {
  const float* x    = (const float*)d_in[0];   // (B, N, C_IN)
  const float* sup  = (const float*)d_in[1];   // (S, N, N)
  const float* W    = (const float*)d_in[2];   // (S, C_OUT, C_IN)
  const float* bias = (const float*)d_in[3];   // (C_OUT,)
  float* out = (float*)d_out;                  // (B, N, C_OUT)
  uint32_t* xp = (uint32_t*)d_ws;              // 4 MB of packed bf16 x frags

  // pack x: 4096 fragments x 32 lanes
  int pack_threads = B_ * KT_ * 2 * 32;        // 131072
  pack_x_kernel<<<pack_threads / 256, 256, 0, stream>>>(x, xp);

  // main: 512 row tiles, 8 waves (256 threads) per block
  graphconv_kernel<<<(N_ / 16) / 8, 256, 0, stream>>>(sup, xp, W, bias, out);
}
